// AttentionBlock_33947421508027
// MI455X (gfx1250) — compile-verified, matrix operand order checked by reading, not run on hardware
//
#include <hip/hip_runtime.h>
#include <hip/hip_bf16.h>

// ---------- shapes (fixed by the reference) ----------
#define B 4
#define C 256
#define HW 4096
#define HEADS 2
#define D 128            // C / HEADS
#define GROUPS 8
#define CPG 32           // C / GROUPS
#define EPS 1e-5f
#define QSCALE 0.08838834764831845f  // 1/sqrt(128)

typedef __attribute__((ext_vector_type(16))) __bf16 v16bf;
typedef __attribute__((ext_vector_type(2)))  __bf16 v2bf;
typedef __attribute__((ext_vector_type(8)))  float  v8f;
typedef __attribute__((ext_vector_type(4)))  unsigned int u32x4;

union BFrag { unsigned int u[8]; u32x4 q[2]; v16bf v; };

__device__ __forceinline__ unsigned short f2bf(float f) {
    unsigned int x = __builtin_bit_cast(unsigned int, f);
    return (unsigned short)((x + 0x7FFFu + ((x >> 16) & 1u)) >> 16);  // RNE
}

#if defined(__has_builtin) && __has_builtin(__builtin_amdgcn_cvt_pk_bf16_f32)
__device__ __forceinline__ unsigned int pack2bf(float lo, float hi) {
    v2bf r = __builtin_amdgcn_cvt_pk_bf16_f32(lo, hi);
    return __builtin_bit_cast(unsigned int, r);
}
#else
__device__ __forceinline__ unsigned int pack2bf(float lo, float hi) {
    return (unsigned int)f2bf(lo) | ((unsigned int)f2bf(hi) << 16);
}
#endif

__device__ __forceinline__ v8f wmma_bf16(const BFrag& a, const BFrag& b, v8f c) {
    return __builtin_amdgcn_wmma_f32_16x16x32_bf16(false, a.v, false, b.v,
                                                   (short)0, c, false, false);
}

// Fragment (16-bit A/B, 16x32 per lane-half layout): dword indices inside a
// 32-wide K chunk are [4h..4h+3] and [8+4h..8+4h+3]  ->  two b128 loads.
__device__ __forceinline__ void load_frag(BFrag& f, const unsigned int* base, int h) {
    f.q[0] = *(const u32x4*)(base + 4 * h);
    f.q[1] = *(const u32x4*)(base + 8 + 4 * h);
}

// Async global -> LDS copy, 16 bytes per lane (ASYNCcnt-tracked).
// Generic shared-pointer low 32 bits ARE the LDS byte offset (ISA 10.2).
__device__ __forceinline__ void async_b128(void* lds_ptr, const void* gaddr) {
    unsigned int lds_off = (unsigned int)(size_t)lds_ptr;
    unsigned long long ga = (unsigned long long)(size_t)gaddr;
    asm volatile("global_load_async_to_lds_b128 %0, %1, off"
                 :: "v"(lds_off), "v"(ga) : "memory");
}

// ---------------------------------------------------------------------------
// Kernel 0: pack fp32 weights -> bf16 (once; removes converts from GEMM loops)
// ---------------------------------------------------------------------------
__global__ void pack_w(const float* __restrict__ qw, const float* __restrict__ pw,
                       unsigned int* __restrict__ qwb, unsigned int* __restrict__ pwb) {
    const int NQ = (3 * C * C) / 2;                 // 98304 dwords
    int idx = blockIdx.x * 256 + threadIdx.x;       // over NQ + 32768
    if (idx < NQ) {
        qwb[idx] = pack2bf(qw[2 * idx], qw[2 * idx + 1]);
    } else {
        int j = idx - NQ;
        pwb[j] = pack2bf(pw[2 * j], pw[2 * j + 1]);
    }
}

// ---------------------------------------------------------------------------
// Kernel 1: per-(batch, group) mean / rstd
// ---------------------------------------------------------------------------
__global__ void gn_stats(const float* __restrict__ x, float* __restrict__ stats) {
    const int bg = blockIdx.x;                       // b*GROUPS + g
    const float* xp = x + (size_t)bg * (CPG * HW);   // group channels contiguous
    float s = 0.f, ss = 0.f;
    for (int i = threadIdx.x; i < CPG * HW; i += 256) {
        float v = xp[i]; s += v; ss += v * v;
    }
    __shared__ float red[512];
    red[threadIdx.x] = s; red[256 + threadIdx.x] = ss;
    __syncthreads();
    for (int st = 128; st > 0; st >>= 1) {
        if (threadIdx.x < st) {
            red[threadIdx.x]       += red[threadIdx.x + st];
            red[256 + threadIdx.x] += red[256 + threadIdx.x + st];
        }
        __syncthreads();
    }
    if (threadIdx.x == 0) {
        const float inv = 1.f / (float)(CPG * HW);
        float mean = red[0] * inv;
        float var  = red[256] * inv - mean * mean;
        stats[2 * bg]     = mean;
        stats[2 * bg + 1] = rsqrtf(var + EPS);
    }
}

// ---------------------------------------------------------------------------
// Kernel 2: apply GN, write x_norm transposed [b][p][c] bf16, packed dwords
// ---------------------------------------------------------------------------
__global__ void gn_apply(const float* __restrict__ x, const float* __restrict__ stats,
                         const float* __restrict__ gamma, const float* __restrict__ beta,
                         unsigned int* __restrict__ xnT) {
    size_t idx = (size_t)blockIdx.x * 256 + threadIdx.x;   // over B * C/2 * HW
    int p = (int)(idx & (HW - 1));
    size_t t = idx >> 12;
    int cp = (int)(t & (C / 2 - 1));
    int b  = (int)(t >> 7);
    int c  = cp * 2;
    int g  = c >> 5;
    float mean = stats[2 * (b * GROUPS + g)];
    float rstd = stats[2 * (b * GROUPS + g) + 1];
    size_t xb = ((size_t)b * C + c) * HW + p;
    float v0 = (x[xb]      - mean) * rstd * gamma[c]     + beta[c];
    float v1 = (x[xb + HW] - mean) * rstd * gamma[c + 1] + beta[c + 1];
    xnT[((size_t)b * HW + p) * (C / 2) + cp] = pack2bf(v0, v1);
}

// ---------------------------------------------------------------------------
// Kernel 3: QKV GEMM.  qkv[o,p] = sum_c W[o,c]*xn[c,p] + bias[o]
//   writes qT [b][h][p][d] (pre-scaled), kT [b][h][p][d], v [b][h][d][p]
// ---------------------------------------------------------------------------
__global__ void qkv_gemm(const unsigned int* __restrict__ xnT,
                         const unsigned int* __restrict__ wb,
                         const float* __restrict__ bias,
                         unsigned short* __restrict__ qT, unsigned short* __restrict__ kT,
                         unsigned short* __restrict__ vmat) {
    int wave = blockIdx.x * 8 + (threadIdx.x >> 5);
    int po = wave % (HW / 16); wave /= (HW / 16);
    int oo = wave % 48;        int b = wave / 48;     // 768/16 = 48 o-tiles
    const int o0 = oo * 16, p0 = po * 16;
    const int lane = threadIdx.x & 31, ml = lane & 15, h = lane >> 4;

    const unsigned int* arow = wb + (size_t)(o0 + ml) * (C / 2);               // A rows = o
    const unsigned int* brow = xnT + ((size_t)b * HW + p0 + ml) * (C / 2);     // B rows = p

    v8f acc = {};
    #pragma unroll
    for (int kc = 0; kc < C; kc += 32) {
        BFrag a, bf;
        load_frag(a,  arow + (kc >> 1), h);
        load_frag(bf, brow + (kc >> 1), h);
        acc = wmma_bf16(a, bf, acc);
    }
    #pragma unroll
    for (int r = 0; r < 8; ++r) {
        int o = o0 + r + 8 * h;
        int p = p0 + ml;
        float val = acc[r] + bias[o];
        int which = o >> 8, rem = o & 255, head = rem >> 7, dd = rem & 127;
        size_t bh = (size_t)b * HEADS + head;
        if (which == 0)
            qT[(bh * HW + p) * D + dd] = f2bf(val * QSCALE);
        else if (which == 1)
            kT[(bh * HW + p) * D + dd] = f2bf(val);
        else
            vmat[(bh * D + dd) * HW + p] = f2bf(val);
    }
}

// ---------------------------------------------------------------------------
// Kernel 4: flash attention.  grid = B*HEADS*32 blocks, 8 waves x 16 rows.
//   K/V tiles staged cooperatively into LDS via global_load_async_to_lds_b128
//   (double-buffered, ASYNCcnt), fragments fed from LDS via ds_load_b128.
// ---------------------------------------------------------------------------
__global__ void flash_attn(const unsigned short* __restrict__ qT,
                           const unsigned short* __restrict__ kT,
                           const unsigned short* __restrict__ vmat,
                           unsigned short* __restrict__ aT) {
    int blk = blockIdx.x;
    int ib = blk & 31; int t = blk >> 5; int head = t & 1; int b = t >> 1;
    const int w = threadIdx.x >> 5, lane = threadIdx.x & 31, tid = threadIdx.x;
    const int ml = lane & 15, h = lane >> 4;
    const int i0 = ib * 128 + w * 16;
    const size_t bh = (size_t)b * HEADS + head;

    const unsigned int* qp = (const unsigned int*)qT + (bh * HW + i0 + ml) * (D / 2);
    const char* kbase = (const char*)kT + bh * HW * (D * 2);     // rows: j, 256 B
    const char* vbase = (const char*)vmat + bh * D * (HW * 2);   // rows: d, 8192 B

    // K tile [32 j][128 d] @ 272 B rows (aligned b128, uniform 2-way banks)
    // V tile [128 d][32 j] @ 80 B rows  (conflict-free)
    __shared__ unsigned int kbuf[2][32][68];
    __shared__ unsigned int vbuf[2][128][20];
    __shared__ unsigned short ptile[8][16][40];   // P: 16x32 bf16 @ 80 B rows

    BFrag qf[4];
    #pragma unroll
    for (int f = 0; f < 4; ++f) load_frag(qf[f], qp + f * 16, h);

    v8f o_acc[8];
    #pragma unroll
    for (int g = 0; g < 8; ++g) o_acc[g] = (v8f){};
    float mrow[8], lrow[8];
    #pragma unroll
    for (int r = 0; r < 8; ++r) { mrow[r] = -__builtin_inff(); lrow[r] = 0.f; }

    // stage tile jt into buffer nb: 4 async b128 instructions per wave
    auto stage = [&](int jt, int nb) {
        #pragma unroll
        for (int cch = 0; cch < 2; ++cch) {          // K: 512 chunks of 16 B
            int id = tid + cch * 256;
            int row = id >> 4, part = id & 15;
            async_b128(&kbuf[nb][row][part * 4],
                       kbase + ((size_t)jt * 32 + row) * (D * 2) + part * 16);
        }
        #pragma unroll
        for (int cch = 0; cch < 2; ++cch) {          // V: 512 chunks of 16 B
            int id = tid + cch * 256;
            int row = id >> 2, part = id & 3;
            async_b128(&vbuf[nb][row][part * 4],
                       vbase + (size_t)row * (HW * 2) + (size_t)jt * 64 + part * 16);
        }
    };

    stage(0, 0);
    const int NT = HW / 32;                          // 128 j-tiles
    for (int jt = 0; jt < NT; ++jt) {
        const int cur = jt & 1;
        __syncthreads();                             // buffer cur^1 free to overwrite
        if (jt + 1 < NT) {
            stage(jt + 1, cur ^ 1);
            asm volatile("s_wait_asynccnt 0x4" ::: "memory");  // tile jt landed
        } else {
            asm volatile("s_wait_asynccnt 0x0" ::: "memory");
        }
        __syncthreads();                             // publish tile jt block-wide

        // ---- S = Qt K for two n-halves of the 32-wide j tile ----
        v8f s[2];
        #pragma unroll
        for (int nh = 0; nh < 2; ++nh) {
            v8f a = (v8f){};
            #pragma unroll
            for (int f = 0; f < 4; ++f) {
                BFrag kb;
                load_frag(kb, &kbuf[cur][nh * 16 + ml][f * 16], h);
                a = wmma_bf16(qf[f], kb, a);
            }
            s[nh] = a;
        }
        // ---- online softmax: row stats over N (lanes within 16-lane half) ----
        float rmax[8];
        #pragma unroll
        for (int r = 0; r < 8; ++r) rmax[r] = fmaxf(s[0][r], s[1][r]);
        #pragma unroll
        for (int mask = 1; mask <= 8; mask <<= 1)
            #pragma unroll
            for (int r = 0; r < 8; ++r)
                rmax[r] = fmaxf(rmax[r], __shfl_xor(rmax[r], mask, 32));
        float alpha[8];
        #pragma unroll
        for (int r = 0; r < 8; ++r) {
            float mn = fmaxf(mrow[r], rmax[r]);
            alpha[r] = __expf(mrow[r] - mn);
            mrow[r] = mn;
        }
        float psum[8] = {0.f,0.f,0.f,0.f,0.f,0.f,0.f,0.f};
        #pragma unroll
        for (int nh = 0; nh < 2; ++nh)
            #pragma unroll
            for (int r = 0; r < 8; ++r) {
                float p = __expf(s[nh][r] - mrow[r]);
                psum[r] += p;
                ptile[w][r + 8 * h][nh * 16 + ml] = f2bf(p);
            }
        #pragma unroll
        for (int mask = 1; mask <= 8; mask <<= 1)
            #pragma unroll
            for (int r = 0; r < 8; ++r)
                psum[r] += __shfl_xor(psum[r], mask, 32);
        #pragma unroll
        for (int r = 0; r < 8; ++r) lrow[r] = lrow[r] * alpha[r] + psum[r];
        #pragma unroll
        for (int g = 0; g < 8; ++g)
            #pragma unroll
            for (int r = 0; r < 8; ++r) o_acc[g][r] *= alpha[r];
        // ---- C-layout -> A-layout via LDS (same-wave DS ops are in-order) ----
        BFrag pf;
        load_frag(pf, (const unsigned int*)&ptile[w][ml][0], h);
        // ---- O += P x Vt (V fragments from LDS) ----
        #pragma unroll
        for (int g = 0; g < 8; ++g) {
            BFrag vb;
            load_frag(vb, &vbuf[cur][g * 16 + ml][0], h);
            o_acc[g] = wmma_bf16(pf, vb, o_acc[g]);
        }
    }
    // epilogue: aT[b][p=i][c = head*128 + g*16 + n] = O / l
    #pragma unroll
    for (int g = 0; g < 8; ++g)
        #pragma unroll
        for (int r = 0; r < 8; ++r) {
            int i = i0 + r + 8 * h;
            int cc = head * D + g * 16 + ml;
            aT[((size_t)b * HW + i) * C + cc] = f2bf(o_acc[g][r] / lrow[r]);
        }
}

// ---------------------------------------------------------------------------
// Kernel 5: proj GEMM + bias + residual (fp32 out)
// ---------------------------------------------------------------------------
__global__ void proj_gemm(const unsigned int* __restrict__ aT,
                          const unsigned int* __restrict__ wb,
                          const float* __restrict__ bias,
                          const float* __restrict__ x, float* __restrict__ out) {
    int wave = blockIdx.x * 8 + (threadIdx.x >> 5);
    int po = wave % (HW / 16); wave /= (HW / 16);
    int oo = wave % (C / 16);  int b = wave / (C / 16);
    const int o0 = oo * 16, p0 = po * 16;
    const int lane = threadIdx.x & 31, ml = lane & 15, h = lane >> 4;

    const unsigned int* arow = wb + (size_t)(o0 + ml) * (C / 2);
    const unsigned int* brow = aT + ((size_t)b * HW + p0 + ml) * (C / 2);

    v8f acc = {};
    #pragma unroll
    for (int kc = 0; kc < C; kc += 32) {
        BFrag a, bf;
        load_frag(a,  arow + (kc >> 1), h);
        load_frag(bf, brow + (kc >> 1), h);
        acc = wmma_bf16(a, bf, acc);
    }
    #pragma unroll
    for (int r = 0; r < 8; ++r) {
        int o = o0 + r + 8 * h;
        int p = p0 + ml;
        size_t idx = ((size_t)b * C + o) * HW + p;
        out[idx] = acc[r] + bias[o] + x[idx];
    }
}

// ---------------------------------------------------------------------------
extern "C" void kernel_launch(void* const* d_in, const int* in_sizes, int n_in,
                              void* d_out, int out_size, void* d_ws, size_t ws_size,
                              hipStream_t stream) {
    const float* x       = (const float*)d_in[0];
    const float* gn_g    = (const float*)d_in[1];
    const float* gn_b    = (const float*)d_in[2];
    const float* qkv_w   = (const float*)d_in[3];
    const float* qkv_b   = (const float*)d_in[4];
    const float* proj_w  = (const float*)d_in[5];
    const float* proj_b  = (const float*)d_in[6];
    float* out = (float*)d_out;

    char* ws = (char*)d_ws;
    const size_t BF_PLANE = (size_t)B * C * HW * sizeof(unsigned short);  // 8 MB
    float*          stats = (float*)ws;                      // 256 B (pad 512)
    unsigned short* xnT   = (unsigned short*)(ws + 512);
    unsigned short* qT    = (unsigned short*)(ws + 512 + 1 * BF_PLANE);
    unsigned short* kT    = (unsigned short*)(ws + 512 + 2 * BF_PLANE);
    unsigned short* vmat  = (unsigned short*)(ws + 512 + 3 * BF_PLANE);
    unsigned short* aT    = (unsigned short*)(ws + 512 + 4 * BF_PLANE);
    unsigned int*   qwb   = (unsigned int*)(ws + 512 + 5 * BF_PLANE);          // 768x256 bf16
    unsigned int*   pwb   = (unsigned int*)(ws + 512 + 5 * BF_PLANE + 393216); // 256x256 bf16

    gn_stats <<<B * GROUPS, 256, 0, stream>>>(x, stats);
    gn_apply <<<(B * (C / 2) * HW) / 256, 256, 0, stream>>>(x, stats, gn_g, gn_b,
                                                            (unsigned int*)xnT);
    pack_w   <<<((3 * C * C + C * C) / 2) / 256, 256, 0, stream>>>(qkv_w, proj_w, qwb, pwb);
    qkv_gemm <<<(48 * (HW / 16) * B) / 8, 256, 0, stream>>>((const unsigned int*)xnT, qwb,
                                                            qkv_b, qT, kT, vmat);
    flash_attn<<<B * HEADS * 32, 256, 0, stream>>>(qT, kT, vmat, aT);
    proj_gemm<<<((C / 16) * (HW / 16) * B) / 8, 256, 0, stream>>>((const unsigned int*)aT, pwb,
                                                                  proj_b, x, out);
}
